// GATWithAutoencoderResidual_26405458936376
// MI455X (gfx1250) — compile-verified, matrix operand order checked by reading, not run on hardware
//
#include <hip/hip_runtime.h>

typedef __bf16    v16bf __attribute__((ext_vector_type(16)));
typedef float     v8f   __attribute__((ext_vector_type(8)));
typedef unsigned  uv4   __attribute__((ext_vector_type(4)));
typedef unsigned short ushort_t;

#define NNODES 50000
#define DIN    512
#define DENC   256
#define H1     4
#define C1     32
#define F1D    128      // H1*C1
#define DOUT   40
#define DOUTP  48       // padded to multiple of 16
#define SLOPE  0.2f

static __device__ __forceinline__ ushort_t f2bf(float f) {
  unsigned u = __float_as_uint(f);
  u += 0x7FFFu + ((u >> 16) & 1u);       // round-to-nearest-even
  return (ushort_t)(u >> 16);
}
// order-preserving float -> u32 key (for exact atomic segment-max)
static __device__ __forceinline__ unsigned fkey(float f) {
  unsigned u = __float_as_uint(f);
  return (u & 0x80000000u) ? ~u : (u | 0x80000000u);
}
static __device__ __forceinline__ float funkey(unsigned k) {
  unsigned u = (k & 0x80000000u) ? (k & 0x7FFFFFFFu) : ~k;
  return __uint_as_float(u);
}

union Frag { uv4 q[2]; v16bf v; };

// ---------------- prep kernels ----------------
__global__ void k_f2bf(const float* __restrict__ s, ushort_t* __restrict__ d, size_t n) {
  size_t t = (size_t)blockIdx.x * blockDim.x + threadIdx.x;
  if (t < n) d[t] = f2bf(s[t]);
}
// W[K x Nn] row-major fp32  ->  Wp[Npad x K] col-major bf16 (zero-padded cols)
__global__ void k_packw(const float* __restrict__ W, ushort_t* __restrict__ Wp,
                        int K, int Nn, int Npad) {
  int t = blockIdx.x * blockDim.x + threadIdx.x;
  if (t >= Npad * K) return;
  int n = t / K, k = t - n * K;
  float v = (n < Nn) ? W[(size_t)k * Nn + n] : 0.0f;
  Wp[t] = f2bf(v);
}
__global__ void k_zero(unsigned* __restrict__ p, size_t n) {
  size_t t = (size_t)blockIdx.x * blockDim.x + threadIdx.x;
  if (t < n) p[t] = 0u;
}

// ---- WMMA bf16 GEMM: C[MxNn] = A[MxK] * B(col-major packed [Npad x K]) ----
// Wave tile: 16 rows x (NT*16) cols; one A fragment feeds NT WMMAs.
// Quad-buffered software pipeline, load-ahead distance 3 bursts: each set's
// registers are rewritten 1 WMMA-burst after their last read (no WAR copies),
// and ~3 bursts of loads remain in flight during every WMMA burst.
template <int NT>
static __device__ __forceinline__ void load_step(const ushort_t* __restrict__ Ap,
                                                 const ushort_t* __restrict__ Bp,
                                                 int K, int k0, Frag& a, Frag (&b)[NT]) {
  a.q[0] = *(const uv4*)(Ap + k0);
  a.q[1] = *(const uv4*)(Ap + k0 + 16);
  #pragma unroll
  for (int t = 0; t < NT; ++t) {
    const ushort_t* bp = Bp + (size_t)t * 16 * K + k0;
    b[t].q[0] = *(const uv4*)(bp);
    b[t].q[1] = *(const uv4*)(bp + 16);
  }
}
template <int NT>
static __device__ __forceinline__ void mma_step(const Frag& a, const Frag (&b)[NT],
                                                v8f (&acc)[NT]) {
  #pragma unroll
  for (int t = 0; t < NT; ++t)
    acc[t] = __builtin_amdgcn_wmma_f32_16x16x32_bf16(false, a.v, false, b[t].v,
                                                     (short)0, acc[t], false, false);
}

template <int NT>
__global__ __launch_bounds__(256)
void k_gemm_bf16(const ushort_t* __restrict__ A, const ushort_t* __restrict__ B,
                 const float* __restrict__ bias,         // nullable
                 float* __restrict__ Cf,                 // fp32 out (or null)
                 ushort_t* __restrict__ Cb,              // bf16 out (or null)
                 int M, int Nn, int K, int relu) {
  const int wave = threadIdx.x >> 5;
  const int lane = threadIdx.x & 31;
  const int half = lane >> 4;
  const int l16  = lane & 15;
  const int row0 = blockIdx.x * 128 + wave * 16;
  const int col0 = blockIdx.y * (16 * NT);

  int arow = row0 + l16; if (arow >= M) arow = M - 1;   // clamp for safe loads
  // lanes 0-15: K {k0..k0+7, k0+16..k0+23}; lanes 16-31: +8 (ISA 16-bit layout)
  const ushort_t* Ap = A + (size_t)arow * K + half * 8;
  const ushort_t* Bp = B + (size_t)(col0 + l16) * K + half * 8;  // padded cols valid

  v8f acc[NT];
  #pragma unroll
  for (int t = 0; t < NT; ++t) acc[t] = (v8f){0.f,0.f,0.f,0.f,0.f,0.f,0.f,0.f};

  // S = K/32 is a multiple of 4 and >= 4 for all call sites (K in {512,256,128}).
  const int S = K >> 5;
  Frag a[4], b[4][NT];
  load_step<NT>(Ap, Bp, K, 0,  a[0], b[0]);
  load_step<NT>(Ap, Bp, K, 32, a[1], b[1]);
  load_step<NT>(Ap, Bp, K, 64, a[2], b[2]);
  int i = 0;
  for (; i + 4 < S; i += 4) {
    mma_step<NT>(a[0], b[0], acc); load_step<NT>(Ap, Bp, K, (i + 3) * 32, a[3], b[3]);
    mma_step<NT>(a[1], b[1], acc); load_step<NT>(Ap, Bp, K, (i + 4) * 32, a[0], b[0]);
    mma_step<NT>(a[2], b[2], acc); load_step<NT>(Ap, Bp, K, (i + 5) * 32, a[1], b[1]);
    mma_step<NT>(a[3], b[3], acc); load_step<NT>(Ap, Bp, K, (i + 6) * 32, a[2], b[2]);
  }
  mma_step<NT>(a[0], b[0], acc); load_step<NT>(Ap, Bp, K, (i + 3) * 32, a[3], b[3]);
  mma_step<NT>(a[1], b[1], acc);
  mma_step<NT>(a[2], b[2], acc);
  mma_step<NT>(a[3], b[3], acc);

  #pragma unroll
  for (int t = 0; t < NT; ++t) {
    const int ocol = col0 + t * 16 + l16;
    if (ocol < Nn) {
      const float bv = bias ? bias[ocol] : 0.0f;
      #pragma unroll
      for (int r = 0; r < 8; ++r) {
        const int orow = row0 + half * 8 + r;   // D layout: VGPR r -> M = r + 8*half
        if (orow < M) {
          float v = acc[t][r] + bv;
          if (relu) v = fmaxf(v, 0.0f);
          const size_t idx = (size_t)orow * Nn + ocol;
          if (Cb) Cb[idx] = f2bf(v); else Cf[idx] = v;
        }
      }
    }
  }
}

// ---------------- GAT attention kernels ----------------
__global__ void k_coeff(const float* __restrict__ hmat, const float* __restrict__ as,
                        const float* __restrict__ ad, float* __restrict__ asrc,
                        float* __restrict__ adst, int n, int H, int C) {
  int t = blockIdx.x * blockDim.x + threadIdx.x;
  if (t >= n * H) return;
  int node = t / H, h = t - node * H;
  const float* row = hmat + (size_t)node * H * C + h * C;
  const float* a1 = as + h * C;
  const float* a2 = ad + h * C;
  float s1 = 0.f, s2 = 0.f;
  for (int c = 0; c < C; ++c) { float v = row[c]; s1 += v * a1[c]; s2 += v * a2[c]; }
  asrc[t] = s1; adst[t] = s2;
}

__global__ void k_edge_max(const int* __restrict__ ei, int E, int ET, int H,
                           const float* __restrict__ asrc, const float* __restrict__ adst,
                           unsigned* __restrict__ m) {
  size_t t = (size_t)blockIdx.x * blockDim.x + threadIdx.x;
  if (t >= (size_t)ET * H) return;
  int eid = (int)(t / H), h = (int)(t - (size_t)eid * H);
  int s, d;
  if (eid < E) { s = ei[eid]; d = ei[E + eid]; } else { s = d = eid - E; }
  float e = asrc[(size_t)s * H + h] + adst[(size_t)d * H + h];
  e = e > 0.f ? e : SLOPE * e;
  atomicMax(&m[(size_t)d * H + h], fkey(e));
}

__global__ void k_edge_exp(const int* __restrict__ ei, int E, int ET, int H,
                           const float* __restrict__ asrc, const float* __restrict__ adst,
                           const unsigned* __restrict__ m, float* __restrict__ ee,
                           float* __restrict__ den) {
  size_t t = (size_t)blockIdx.x * blockDim.x + threadIdx.x;
  if (t >= (size_t)ET * H) return;
  int eid = (int)(t / H), h = (int)(t - (size_t)eid * H);
  int s, d;
  if (eid < E) { s = ei[eid]; d = ei[E + eid]; } else { s = d = eid - E; }
  float e = asrc[(size_t)s * H + h] + adst[(size_t)d * H + h];
  e = e > 0.f ? e : SLOPE * e;
  float v = expf(e - funkey(m[(size_t)d * H + h]));
  ee[t] = v;
  atomicAdd(&den[(size_t)d * H + h], v);
}

__global__ void k_edge_aggr(const int* __restrict__ ei, int E, int ET, int H, int C,
                            const float* __restrict__ hmat, const float* __restrict__ ee,
                            const float* __restrict__ den, float* __restrict__ outb) {
  const int F = H * C;
  size_t t = (size_t)blockIdx.x * blockDim.x + threadIdx.x;
  if (t >= (size_t)ET * F) return;
  int eid = (int)(t / F), f = (int)(t - (size_t)eid * F), h = f / C;
  int s, d;
  if (eid < E) { s = ei[eid]; d = ei[E + eid]; } else { s = d = eid - E; }
  float alpha = ee[(size_t)eid * H + h] / fmaxf(den[(size_t)d * H + h], 1e-16f);
  atomicAdd(&outb[(size_t)d * F + f], hmat[(size_t)s * F + f] * alpha);
}

// h = relu(agg + bias[f] + res), stored bf16
__global__ void k_combine(const float* __restrict__ agg, const float* __restrict__ bias,
                          const float* __restrict__ res, ushort_t* __restrict__ outb,
                          size_t total, int F) {
  size_t t = (size_t)blockIdx.x * blockDim.x + threadIdx.x;
  if (t >= total) return;
  int f = (int)(t % F);
  outb[t] = f2bf(fmaxf(agg[t] + bias[f] + res[t], 0.0f));
}

__global__ void k_final(const float* __restrict__ agg, const float* __restrict__ bias,
                        const float* __restrict__ res, float* __restrict__ out,
                        size_t total, int F) {
  size_t t = (size_t)blockIdx.x * blockDim.x + threadIdx.x;
  if (t >= total) return;
  int f = (int)(t % F);
  out[t] = agg[t] + bias[f] + res[t];
}

// ---------------- host side ----------------
static inline unsigned cdiv(size_t a, size_t b) { return (unsigned)((a + b - 1) / b); }

extern "C" void kernel_launch(void* const* d_in, const int* in_sizes, int n_in,
                              void* d_out, int out_size, void* d_ws, size_t ws_size,
                              hipStream_t stream) {
  (void)n_in; (void)out_size; (void)ws_size;
  const float* x       = (const float*)d_in[0];
  const int*   ei      = (const int*)d_in[1];
  const float* ae_w1   = (const float*)d_in[2];
  const float* ae_b1   = (const float*)d_in[3];
  const float* ae_w2   = (const float*)d_in[4];
  const float* ae_b2   = (const float*)d_in[5];
  const float* gat1_w  = (const float*)d_in[6];
  const float* gat1_as = (const float*)d_in[7];
  const float* gat1_ad = (const float*)d_in[8];
  const float* gat1_b  = (const float*)d_in[9];
  const float* gat2_w  = (const float*)d_in[10];
  const float* gat2_as = (const float*)d_in[11];
  const float* gat2_ad = (const float*)d_in[12];
  const float* gat2_b  = (const float*)d_in[13];
  const float* res1_w  = (const float*)d_in[14];
  const float* res1_b  = (const float*)d_in[15];
  const float* res2_w  = (const float*)d_in[16];
  const float* res2_b  = (const float*)d_in[17];

  const int E  = in_sizes[1] / 2;
  const int ET = E + NNODES;

  // -------- workspace arena with lifetime-based region reuse --------
  char* base = (char*)d_ws;
  const size_t szA = 51200000, szB = 25600000, szC = 25600000, szD = 25600000, szE = 25600000;
  char* rA = base;            // xBf | later: out1, hBf, hg2
  char* rB = rA + szA;        // xe1Bf | later: ee1, asrc1, adst1, m1, den1
  char* rC = rB + szB;        // xeBf  | later: r2, out2, ee2, asrc2..den2
  char* rD = rC + szC;        // hg1
  char* rE = rD + szD;        // r1
  char* rW = rE + szE;        // packed weights

  ushort_t* xBf   = (ushort_t*)rA;
  float*    out1  = (float*)rA;
  ushort_t* hBf   = (ushort_t*)(rA + 25600000);
  float*    hg2   = (float*)(rA + 25600000 + 12800000);

  ushort_t* xe1Bf = (ushort_t*)rB;
  float*    ee1   = (float*)rB;
  size_t ee1B = (((size_t)ET * H1 * 4) + 255) & ~(size_t)255;
  float*    asrc1 = (float*)(rB + ee1B);
  float*    adst1 = asrc1 + (size_t)NNODES * H1;
  unsigned* m1    = (unsigned*)(adst1 + (size_t)NNODES * H1);
  float*    den1  = (float*)(m1 + (size_t)NNODES * H1);

  ushort_t* xeBf  = (ushort_t*)rC;
  float*    r2    = (float*)rC;
  float*    out2  = (float*)(rC + 8000000);
  float*    ee2   = (float*)(rC + 16000000);
  size_t ee2B = (((size_t)ET * 4) + 255) & ~(size_t)255;
  float*    asrc2 = (float*)(rC + 16000000 + ee2B);
  float*    adst2 = asrc2 + NNODES;
  unsigned* m2    = (unsigned*)(adst2 + NNODES);
  float*    den2  = (float*)(m2 + NNODES);

  float* hg1 = (float*)rD;
  float* r1  = (float*)rE;

  ushort_t* w1p  = (ushort_t*)rW;                 // 512*256
  ushort_t* w2p  = w1p + (size_t)DIN * DENC;      // 256*256
  ushort_t* g1wp = w2p + (size_t)DENC * DENC;     // 256*128
  ushort_t* r1wp = g1wp + (size_t)DENC * F1D;     // 256*128
  ushort_t* g2wp = r1wp + (size_t)DENC * F1D;     // 128*48
  ushort_t* r2wp = g2wp + (size_t)F1D * DOUTP;    // 128*48

  const dim3 blk(256);

  // ---- prep: convert x, pack weights ----
  k_f2bf<<<cdiv((size_t)NNODES * DIN, 256), blk, 0, stream>>>(x, xBf, (size_t)NNODES * DIN);
  k_packw<<<cdiv((size_t)DENC * DIN, 256), blk, 0, stream>>>(ae_w1, w1p, DIN, DENC, DENC);
  k_packw<<<cdiv((size_t)DENC * DENC, 256), blk, 0, stream>>>(ae_w2, w2p, DENC, DENC, DENC);
  k_packw<<<cdiv((size_t)F1D * DENC, 256), blk, 0, stream>>>(gat1_w, g1wp, DENC, F1D, F1D);
  k_packw<<<cdiv((size_t)F1D * DENC, 256), blk, 0, stream>>>(res1_w, r1wp, DENC, F1D, F1D);
  k_packw<<<cdiv((size_t)DOUTP * F1D, 256), blk, 0, stream>>>(gat2_w, g2wp, F1D, DOUT, DOUTP);
  k_packw<<<cdiv((size_t)DOUTP * F1D, 256), blk, 0, stream>>>(res2_w, r2wp, F1D, DOUT, DOUTP);

  const unsigned gM = cdiv(NNODES, 128);
  // ---- autoencoder: xe1 = relu(x@w1+b1); xe = relu(xe1@w2+b2) ----
  k_gemm_bf16<4><<<dim3(gM, DENC / 64), blk, 0, stream>>>(xBf, w1p, ae_b1, nullptr, xe1Bf,
                                                          NNODES, DENC, DIN, 1);
  k_gemm_bf16<4><<<dim3(gM, DENC / 64), blk, 0, stream>>>(xe1Bf, w2p, ae_b2, nullptr, xeBf,
                                                          NNODES, DENC, DENC, 1);
  // ---- GAT1 projections + residual ----
  k_gemm_bf16<4><<<dim3(gM, F1D / 64), blk, 0, stream>>>(xeBf, g1wp, nullptr, hg1, nullptr,
                                                         NNODES, F1D, DENC, 0);
  k_gemm_bf16<4><<<dim3(gM, F1D / 64), blk, 0, stream>>>(xeBf, r1wp, res1_b, r1, nullptr,
                                                         NNODES, F1D, DENC, 0);
  // ---- GAT1 attention ----
  k_coeff<<<cdiv((size_t)NNODES * H1, 256), blk, 0, stream>>>(hg1, gat1_as, gat1_ad,
                                                              asrc1, adst1, NNODES, H1, C1);
  k_zero<<<cdiv((size_t)NNODES * H1 * 2, 256), blk, 0, stream>>>(m1, (size_t)NNODES * H1 * 2); // m1+den1
  k_zero<<<cdiv((size_t)NNODES * F1D, 256), blk, 0, stream>>>((unsigned*)out1, (size_t)NNODES * F1D);
  k_edge_max<<<cdiv((size_t)ET * H1, 256), blk, 0, stream>>>(ei, E, ET, H1, asrc1, adst1, m1);
  k_edge_exp<<<cdiv((size_t)ET * H1, 256), blk, 0, stream>>>(ei, E, ET, H1, asrc1, adst1, m1, ee1, den1);
  k_edge_aggr<<<cdiv((size_t)ET * F1D, 256), blk, 0, stream>>>(ei, E, ET, H1, C1, hg1, ee1, den1, out1);
  // ---- h = relu(g1 + gat1_b + r1) (bf16) ----
  k_combine<<<cdiv((size_t)NNODES * F1D, 256), blk, 0, stream>>>(out1, gat1_b, r1, hBf,
                                                                 (size_t)NNODES * F1D, F1D);
  // ---- GAT2 projections + residual ----
  k_gemm_bf16<3><<<dim3(gM, 1), blk, 0, stream>>>(hBf, g2wp, nullptr, hg2, nullptr,
                                                  NNODES, DOUT, F1D, 0);
  k_gemm_bf16<3><<<dim3(gM, 1), blk, 0, stream>>>(hBf, r2wp, res2_b, r2, nullptr,
                                                  NNODES, DOUT, F1D, 0);
  // ---- GAT2 attention (1 head) ----
  k_coeff<<<cdiv(NNODES, 256), blk, 0, stream>>>(hg2, gat2_as, gat2_ad, asrc2, adst2, NNODES, 1, DOUT);
  k_zero<<<cdiv((size_t)NNODES * 2, 256), blk, 0, stream>>>(m2, (size_t)NNODES * 2);          // m2+den2
  k_zero<<<cdiv((size_t)NNODES * DOUT, 256), blk, 0, stream>>>((unsigned*)out2, (size_t)NNODES * DOUT);
  k_edge_max<<<cdiv((size_t)ET, 256), blk, 0, stream>>>(ei, E, ET, 1, asrc2, adst2, m2);
  k_edge_exp<<<cdiv((size_t)ET, 256), blk, 0, stream>>>(ei, E, ET, 1, asrc2, adst2, m2, ee2, den2);
  k_edge_aggr<<<cdiv((size_t)ET * DOUT, 256), blk, 0, stream>>>(ei, E, ET, 1, DOUT, hg2, ee2, den2, out2);
  // ---- out = g2 + gat2_b + h@res2_w + res2_b ----
  k_final<<<cdiv((size_t)NNODES * DOUT, 256), blk, 0, stream>>>(out2, gat2_b, r2, (float*)d_out,
                                                               (size_t)NNODES * DOUT, DOUT);
}
